// Disentangle_out_layer_22230750724256
// MI455X (gfx1250) — matrix operand order repaired and checked by reading, not compiled, not run on hardware
//
#include <hip/hip_runtime.h>
#include <hip/hip_bf16.h>

#define NN 8192
#define DD 128
#define JSTRIPS 16

typedef __attribute__((ext_vector_type(16))) _Float16 v16h;
typedef __attribute__((ext_vector_type(8)))  float    v8f;

union Frag {
    v16h v;
    _Float16 h[16];
    uint4 u[2];
};

// ---------------- pass 0: Z -> f16 (row major + transposed), h = 0.5*Z ----
__global__ __launch_bounds__(256) void pass0_convert(
    const float* __restrict__ Z, _Float16* __restrict__ Zh,
    _Float16* __restrict__ ZhT, float* __restrict__ hout) {
    int idx = blockIdx.x * 256 + threadIdx.x;      // over N*D
    int i = idx >> 7;
    int d = idx & 127;
    float z = Z[idx];
    _Float16 zh = (_Float16)z;
    Zh[idx] = zh;
    ZhT[((size_t)d << 13) + i] = zh;               // [D][N]
    hout[idx] = 0.5f * z;                          // base for second GEMM
}

// ---------------- pass 1: rowsum partials via WMMA (16x64 tile per wave) ---
__global__ __launch_bounds__(32) void pass1_rowsum(
    const _Float16* __restrict__ Zh, float* __restrict__ partial) {
    const int lane  = threadIdx.x;
    const int halfg = lane >> 4;
    const int l15   = lane & 15;
    const int i0 = blockIdx.y << 4;                // 512 row tiles
    const int j0 = blockIdx.x << 6;                // 128 col tiles of 64

    v8f acc0 = {}, acc1 = {}, acc2 = {}, acc3 = {};
#pragma unroll
    for (int k0 = 0; k0 < DD; k0 += 32) {
        Frag a;
        const _Float16* ap = Zh + ((size_t)(i0 + l15) << 7) + k0;
        a.u[0] = *(const uint4*)(ap + halfg * 16);
        a.u[1] = *(const uint4*)(ap + halfg * 16 + 8);
        const _Float16* bp = Zh + ((size_t)(j0 + l15) << 7) + k0 + halfg * 16;
        Frag b0, b1, b2, b3;
        b0.u[0] = *(const uint4*)(bp);
        b0.u[1] = *(const uint4*)(bp + 8);
        b1.u[0] = *(const uint4*)(bp + 16 * DD);
        b1.u[1] = *(const uint4*)(bp + 16 * DD + 8);
        b2.u[0] = *(const uint4*)(bp + 32 * DD);
        b2.u[1] = *(const uint4*)(bp + 32 * DD + 8);
        b3.u[0] = *(const uint4*)(bp + 48 * DD);
        b3.u[1] = *(const uint4*)(bp + 48 * DD + 8);
        acc0 = __builtin_amdgcn_wmma_f32_16x16x32_f16(false, a.v, false, b0.v, (short)0, acc0, false, false);
        acc1 = __builtin_amdgcn_wmma_f32_16x16x32_f16(false, a.v, false, b1.v, (short)0, acc1, false, false);
        acc2 = __builtin_amdgcn_wmma_f32_16x16x32_f16(false, a.v, false, b2.v, (short)0, acc2, false, false);
        acc3 = __builtin_amdgcn_wmma_f32_16x16x32_f16(false, a.v, false, b3.v, (short)0, acc3, false, false);
    }
    // exp() then reduce over the 64 columns this wave owns (deterministic).
    float rs[8];
#pragma unroll
    for (int r = 0; r < 8; ++r)
        rs[r] = (__expf(acc0[r]) + __expf(acc1[r])) + (__expf(acc2[r]) + __expf(acc3[r]));
#pragma unroll
    for (int m = 1; m < 16; m <<= 1)
#pragma unroll
        for (int r = 0; r < 8; ++r)
            rs[r] += __shfl_xor(rs[r], m, 32);     // reduce within 16-lane half
    if (l15 == 0) {
#pragma unroll
        for (int r = 0; r < 8; ++r)
            partial[((size_t)(i0 + halfg * 8 + r) << 7) + blockIdx.x] = rs[r];
    }
}

// ---------------- pass 1b: invrs[j] = 1 / sum(partials) ------------------
__global__ __launch_bounds__(256) void pass1b_invrs(
    const float* __restrict__ partial, float* __restrict__ invrs) {
    int i = blockIdx.x * 256 + threadIdx.x;
    const float4* p = (const float4*)(partial + ((size_t)i << 7));
    float s = 0.f;
#pragma unroll
    for (int t = 0; t < 32; ++t) {
        float4 v = p[t];
        s += (v.x + v.y) + (v.z + v.w);
    }
    invrs[i] = 1.0f / s;
}

// ---- pass 2: recompute S tile (WMMA), alpha out, LDS transpose, P*Z ------
__global__ __launch_bounds__(32) void pass2_fused(
    const _Float16* __restrict__ Zh, const _Float16* __restrict__ ZhT,
    const float* __restrict__ adj, const float* __restrict__ invrs,
    float* __restrict__ alpha, float* __restrict__ hout) {
    __shared__ float pshare[2][16][20];            // padded: no bank conflicts
    const int lane  = threadIdx.x;
    const int halfg = lane >> 4;
    const int l15   = lane & 15;
    const int i0   = blockIdx.x << 4;
    const int jbeg = blockIdx.y * (NN / JSTRIPS);

    v8f hacc[8] = {};                              // 16x128 f32 accumulator

    // A fragments of the fixed i-tile (rows i0..i0+15, all of K=128) are
    // invariant across the whole j loop: hoist them (4 x v16h = 32 VGPRs).
    Frag ainv[4];
#pragma unroll
    for (int kk = 0; kk < 4; ++kk) {
        const _Float16* ap = Zh + ((size_t)(i0 + l15) << 7) + kk * 32;
        ainv[kk].u[0] = *(const uint4*)(ap + halfg * 16);
        ainv[kk].u[1] = *(const uint4*)(ap + halfg * 16 + 8);
    }

    for (int j0 = jbeg; j0 < jbeg + NN / JSTRIPS; j0 += 32) {
        // --- prefetch the streaming adj tile (non-temporal) so the HBM
        //     stream overlaps with the WMMA recompute of S ------------------
        float adj0[8], adj1[8];
#pragma unroll
        for (int r = 0; r < 8; ++r) {
            const size_t rowoff = (size_t)(i0 + halfg * 8 + r) << 13;
            adj0[r] = __builtin_nontemporal_load(adj + rowoff + j0 + l15);
            adj1[r] = __builtin_nontemporal_load(adj + rowoff + j0 + 16 + l15);
        }
        const float ir0 = invrs[j0 + l15];
        const float ir1 = invrs[j0 + 16 + l15];

        // --- recompute two 16x16 tiles of S = Z Z^T (bitwise same as pass1)
        v8f s0 = {}, s1 = {};
#pragma unroll
        for (int kk = 0; kk < 4; ++kk) {
            Frag b0, b1;
            const _Float16* bp = Zh + ((size_t)(j0 + l15) << 7) + kk * 32 + halfg * 16;
            b0.u[0] = *(const uint4*)(bp);
            b0.u[1] = *(const uint4*)(bp + 8);
            b1.u[0] = *(const uint4*)(bp + 16 * DD);
            b1.u[1] = *(const uint4*)(bp + 16 * DD + 8);
            s0 = __builtin_amdgcn_wmma_f32_16x16x32_f16(false, ainv[kk].v, false, b0.v, (short)0, s0, false, false);
            s1 = __builtin_amdgcn_wmma_f32_16x16x32_f16(false, ainv[kk].v, false, b1.v, (short)0, s1, false, false);
        }

        // --- alpha = exp(S)/rowsum[j]; stream alpha out (NT); stage p -----
#pragma unroll
        for (int r = 0; r < 8; ++r) {
            const int m = halfg * 8 + r;
            const size_t rowoff = (size_t)(i0 + m) << 13;
            const float a0 = __expf(s0[r]) * ir0;
            const float a1 = __expf(s1[r]) * ir1;
            __builtin_nontemporal_store(a0, alpha + rowoff + j0 + l15);
            __builtin_nontemporal_store(a1, alpha + rowoff + j0 + 16 + l15);
            pshare[0][m][l15] = a0 * adj0[r];
            pshare[1][m][l15] = a1 * adj1[r];
        }
        __syncthreads();
        // --- C-layout -> A-layout transpose via LDS, convert to f16 --------
        Frag af;
#pragma unroll
        for (int e = 0; e < 8; ++e) {
            af.h[e]     = (_Float16)pshare[halfg][l15][e];
            af.h[8 + e] = (_Float16)pshare[halfg][l15][8 + e];
        }
        // --- accumulate H += P(16x32) * Z(32x128) via 8 WMMAs -------------
#pragma unroll
        for (int dt = 0; dt < 8; ++dt) {
            Frag b;
            const _Float16* bp = ZhT + ((size_t)(dt * 16 + l15) << 13) + j0 + halfg * 16;
            b.u[0] = *(const uint4*)(bp);
            b.u[1] = *(const uint4*)(bp + 8);
            hacc[dt] = __builtin_amdgcn_wmma_f32_16x16x32_f16(false, af.v, false, b.v, (short)0, hacc[dt], false, false);
        }
        __syncthreads();
    }
    // --- h += 0.5 * partial H (base 0.5*Z written by pass0) ---------------
#pragma unroll
    for (int dt = 0; dt < 8; ++dt)
#pragma unroll
        for (int r = 0; r < 8; ++r) {
            const int m = halfg * 8 + r;
            atomicAdd(&hout[((size_t)(i0 + m) << 7) + dt * 16 + l15], 0.5f * hacc[dt][r]);
        }
}

extern "C" void kernel_launch(void* const* d_in, const int* in_sizes, int n_in,
                              void* d_out, int out_size, void* d_ws, size_t ws_size,
                              hipStream_t stream) {
    const float* Z   = (const float*)d_in[0];      // [N][D]
    const float* adj = (const float*)d_in[1];      // [N][N]
    float* hout  = (float*)d_out;                          // [N][D]
    float* alpha = (float*)d_out + (size_t)NN * DD;        // [N][N]

    char* ws = (char*)d_ws;
    _Float16* Zh   = (_Float16*)ws;                                     // 2 MB
    _Float16* ZhT  = (_Float16*)(ws + (size_t)NN * DD * 2);             // 2 MB
    float*    partial = (float*)(ws + (size_t)NN * DD * 4);             // 4 MB
    float*    invrs   = (float*)(ws + (size_t)NN * DD * 4 +
                                 (size_t)NN * (NN / 64) * 4);           // 32 KB

    pass0_convert<<<NN * DD / 256, 256, 0, stream>>>(Z, Zh, ZhT, hout);
    pass1_rowsum<<<dim3(NN / 64, NN / 16), 32, 0, stream>>>(Zh, partial);
    pass1b_invrs<<<NN / 256, 256, 0, stream>>>(partial, invrs);
    pass2_fused<<<dim3(NN / 16, JSTRIPS), 32, 0, stream>>>(Zh, ZhT, adj, invrs, alpha, hout);
}
